// Net_62689342652565
// MI455X (gfx1250) — compile-verified, matrix operand order checked by reading, not run on hardware
//
#include <hip/hip_runtime.h>
#include <stdint.h>

// ---------------------------------------------------------------------------
// Problem constants (match reference)
// ---------------------------------------------------------------------------
#define N_NODES   20000
#define N_EDGES   200000
#define NUM_GRAPHS 512
// layer channel configs: (16->48), (48->32), (32->16); edge MLP hidden = 128

typedef _Float16 f16;
typedef __attribute__((ext_vector_type(16))) _Float16 v16h;
typedef __attribute__((ext_vector_type(8)))  _Float16 v8h;
typedef __attribute__((ext_vector_type(8)))  float    v8f;

union FragH {
    v16h v;
    v8h  h8[2];
    _Float16 e[16];
};

// ---------------------------------------------------------------------------
// Pack a row-major f32 weight matrix W[Krows x Nall] into WMMA B-fragment
// order (f16), zero-padding K up to a multiple of 32.
// Layout: out[(((b*kbCnt)+kb)*32 + lane)*16 + j]
//   lane: n = b*16 + (lane&15); g = lane>>4
//   j<8 : K = kb*32 + g*8 + j
//   j>=8: K = kb*32 + 16 + g*8 + (j-8)
// ---------------------------------------------------------------------------
__global__ void pack_b_kernel(const float* __restrict__ W, f16* __restrict__ out,
                              int Krows, int Nall)
{
    const int kbCnt = (Krows + 31) / 32;
    const int total = (Nall / 16) * kbCnt * 512;
    int idx = blockIdx.x * blockDim.x + threadIdx.x;
    if (idx >= total) return;
    const int j    = idx & 15;
    const int lane = (idx >> 4) & 31;
    const int rest = idx >> 9;           // b*kbCnt + kb
    const int kb   = rest % kbCnt;
    const int b    = rest / kbCnt;
    const int g    = lane >> 4;
    const int nl   = lane & 15;
    const int k32  = (j < 8) ? (g * 8 + j) : (16 + g * 8 + (j - 8));
    const int K    = kb * 32 + k32;
    const int n    = b * 16 + nl;
    float val = (K < Krows) ? W[(size_t)K * Nall + n] : 0.0f;
    out[idx] = (f16)val;
}

// ---------------------------------------------------------------------------
// Degree accumulation (for aggr='mean')
// ---------------------------------------------------------------------------
__global__ void deg_kernel(const int* __restrict__ tgt, float* __restrict__ deg, int E)
{
    int e = blockIdx.x * blockDim.x + threadIdx.x;
    if (e < E) atomicAdd(&deg[tgt[e]], 1.0f);
}

// ---------------------------------------------------------------------------
// Fused NNConv edge kernel. One wave32 per tile of 16 edges.
//   h   = relu(ea @ W1 + b1)                        (WMMA, K=16 padded to 32)
//   We  = h @ W2 + b2  (never materialized)
//   msg = x[src] (contracted against We)            (WMMA + VALU scaling)
//   agg[tgt] += msg                                 (f32 global atomics)
// ---------------------------------------------------------------------------
template<int IN_C, int OUT_C>
__global__ void __launch_bounds__(32)
nnconv_kernel(const float* __restrict__ xin,
              const int*   __restrict__ src,
              const int*   __restrict__ tgt,
              const float* __restrict__ ea,
              const f16*   __restrict__ w1p,   // packed [8 nblk][32 lane][16]
              const float* __restrict__ b1,    // [128]
              const f16*   __restrict__ w2p,   // packed [NBLK][4 kb][32 lane][16]
              const float* __restrict__ b2,    // [IN_C*OUT_C]
              float*       __restrict__ agg)   // [N_NODES*OUT_C]
{
    constexpr int NS = OUT_C / 16;           // N-slices of the output tile
    __shared__ __align__(16) _Float16 hsm[16 * 128];
    __shared__ float xs[16 * IN_C];
    __shared__ int   sidx[16];
    __shared__ int   tidx[16];

    const int lane = threadIdx.x;
    const int e0   = blockIdx.x * 16;
    const int m    = lane & 15;
    const int g    = lane >> 4;
    const int nl   = lane & 15;

    if (lane < 16) {
        sidx[lane] = src[e0 + lane];
        tidx[lane] = tgt[e0 + lane];
    }
    __syncthreads();

    // gather x[src] tile into LDS (f32)
    for (int t = lane; t < 16 * IN_C; t += 32) {
        int r = t / IN_C, c = t % IN_C;
        xs[t] = xin[(size_t)sidx[r] * IN_C + c];
    }

    // ---- stage 1: h tile = relu(ea @ W1 + b1), via WMMA with K padded ----
    FragH aEA;
#pragma unroll
    for (int j = 0; j < 8; j++)
        aEA.e[j] = (f16)ea[(size_t)(e0 + m) * 16 + g * 8 + j];
#pragma unroll
    for (int j = 8; j < 16; j++)
        aEA.e[j] = (f16)0.0f;

    for (int nb = 0; nb < 8; nb++) {
        FragH bF;
        const f16* bp = w1p + ((size_t)(nb * 32 + lane)) * 16;
        bF.h8[0] = *(const v8h*)(bp);
        bF.h8[1] = *(const v8h*)(bp + 8);
        v8f cc = {0.f, 0.f, 0.f, 0.f, 0.f, 0.f, 0.f, 0.f};
        cc = __builtin_amdgcn_wmma_f32_16x16x32_f16(
                 false, aEA.v, false, bF.v, (short)0, cc, false, false);
        float bb = b1[nb * 16 + nl];
#pragma unroll
        for (int v = 0; v < 8; v++) {
            float val = cc[v] + bb;
            val = val > 0.f ? val : 0.f;
            hsm[(v + 8 * g) * 128 + nb * 16 + nl] = (f16)val;
        }
    }
    __syncthreads();

    // ---- re-read h tile from LDS in A-fragment order (K blocks of 32) ----
    FragH aF[4];
#pragma unroll
    for (int kb = 0; kb < 4; kb++) {
        aF[kb].h8[0] = *(const v8h*)&hsm[m * 128 + kb * 32 + g * 8];
        aF[kb].h8[1] = *(const v8h*)&hsm[m * 128 + kb * 32 + 16 + g * 8];
    }

    v8f acc[NS];
#pragma unroll
    for (int s = 0; s < NS; s++)
#pragma unroll
        for (int v = 0; v < 8; v++) acc[s][v] = 0.f;

    // ---- stage 2: stream We blocks, contract with x[src] on the fly ----
    for (int i = 0; i < IN_C; i++) {
        float xv[8];
#pragma unroll
        for (int v = 0; v < 8; v++)
            xv[v] = xs[(v + 8 * g) * IN_C + i];
#pragma unroll
        for (int s = 0; s < NS; s++) {
            const int b = i * NS + s;                 // N-block of W2
            const f16* bp = w2p + (size_t)b * 2048 + (size_t)lane * 16;
            v8f cc = {0.f, 0.f, 0.f, 0.f, 0.f, 0.f, 0.f, 0.f};
#pragma unroll
            for (int kb = 0; kb < 4; kb++) {
                FragH bF;
                bF.h8[0] = *(const v8h*)(bp + kb * 512);
                bF.h8[1] = *(const v8h*)(bp + kb * 512 + 8);
                cc = __builtin_amdgcn_wmma_f32_16x16x32_f16(
                         false, aF[kb].v, false, bF.v, (short)0, cc, false, false);
            }
#pragma unroll
            for (int v = 0; v < 8; v++)
                acc[s][v] += xv[v] * cc[v];
        }
    }

    // ---- b2 contribution + scatter (mean-divide happens in node kernel) ----
#pragma unroll
    for (int s = 0; s < NS; s++) {
        const int o = s * 16 + nl;
#pragma unroll
        for (int v = 0; v < 8; v++) {
            const int mm = v + 8 * g;
            float extra = 0.f;
            for (int i = 0; i < IN_C; i++)
                extra += xs[mm * IN_C + i] * b2[i * OUT_C + o];
            atomicAdd(&agg[(size_t)tidx[mm] * OUT_C + o], acc[s][v] + extra);
        }
    }
}

// ---------------------------------------------------------------------------
// Per-node: y = relu(agg/max(deg,1) + x@root + bias)
// ---------------------------------------------------------------------------
template<int IN_C, int OUT_C>
__global__ void node_update_kernel(const float* __restrict__ xin,
                                   const float* __restrict__ agg,
                                   const float* __restrict__ deg,
                                   const float* __restrict__ root,  // [IN_C,OUT_C]
                                   const float* __restrict__ bias,
                                   float*       __restrict__ yout)
{
    int idx = blockIdx.x * blockDim.x + threadIdx.x;
    if (idx >= N_NODES * OUT_C) return;
    int n = idx / OUT_C, o = idx % OUT_C;
    float d = deg[n];
    d = d > 1.0f ? d : 1.0f;
    float v = agg[idx] / d + bias[o];
    const float* xr = xin + (size_t)n * IN_C;
    for (int i = 0; i < IN_C; i++)
        v += xr[i] * root[i * OUT_C + o];
    yout[idx] = v > 0.f ? v : 0.f;
}

// ---------------------------------------------------------------------------
// Set2Set (2 steps) + final MLP. One block (64 threads) per graph; batch is
// sorted, so node range found with binary search.
// ---------------------------------------------------------------------------
__device__ __forceinline__ float sigm(float x) { return 1.0f / (1.0f + expf(-x)); }

__global__ void __launch_bounds__(64)
set2set_kernel(const float* __restrict__ x,      // [N_NODES,16]
               const int*   __restrict__ batch,  // [N_NODES] sorted
               const float* __restrict__ Wih,    // [64,32]
               const float* __restrict__ Whh,    // [64,16]
               const float* __restrict__ bih,    // [64]
               const float* __restrict__ bhh,    // [64]
               const float* __restrict__ l1W, const float* __restrict__ l1b,
               const float* __restrict__ l2W, const float* __restrict__ l2b,
               const float* __restrict__ lfW, const float* __restrict__ lfb,
               float*       __restrict__ out)    // [NUM_GRAPHS]
{
    __shared__ float h[16], c[16], qs[32], gates[64], red[64], rsh[16];
    __shared__ float y1s[16], y2s[8];
    const int tid = threadIdx.x;
    const int gph = blockIdx.x;

    // lower bound binary search
    int beg, end;
    {
        int lo = 0, hi = N_NODES;
        while (lo < hi) { int mid = (lo + hi) >> 1; if (batch[mid] < gph) lo = mid + 1; else hi = mid; }
        beg = lo;
        lo = beg; hi = N_NODES;
        while (lo < hi) { int mid = (lo + hi) >> 1; if (batch[mid] < gph + 1) lo = mid + 1; else hi = mid; }
        end = lo;
    }

    if (tid < 16) { h[tid] = 0.f; c[tid] = 0.f; }
    if (tid < 32) qs[tid] = 0.f;
    __syncthreads();

    for (int step = 0; step < 2; step++) {
        // LSTM cell: gates = qs@Wih.T + bih + h@Whh.T + bhh
        {
            float gv = bih[tid] + bhh[tid];
            for (int k = 0; k < 32; k++) gv += qs[k] * Wih[tid * 32 + k];
            for (int k = 0; k < 16; k++) gv += h[k] * Whh[tid * 16 + k];
            gates[tid] = gv;
        }
        __syncthreads();
        if (tid < 16) {
            float ig = sigm(gates[tid]);
            float fg = sigm(gates[16 + tid]);
            float gg = tanhf(gates[32 + tid]);
            float og = sigm(gates[48 + tid]);
            float cn = fg * c[tid] + ig * gg;
            c[tid] = cn;
            h[tid] = og * tanhf(cn);
        }
        __syncthreads();

        // attention logits max
        float lmax = -3.402823466e38f;
        for (int n = beg + tid; n < end; n += 64) {
            float e = 0.f;
            for (int k = 0; k < 16; k++) e += x[(size_t)n * 16 + k] * h[k];
            lmax = lmax > e ? lmax : e;
        }
        red[tid] = lmax;
        __syncthreads();
        for (int off = 32; off > 0; off >>= 1) {
            if (tid < off) red[tid] = red[tid] > red[tid + off] ? red[tid] : red[tid + off];
            __syncthreads();
        }
        float emax = red[0];
        emax = emax > -1e30f ? emax : -1e30f;
        __syncthreads();

        if (tid == 0) {}
        if (tid < 16) rsh[tid] = 0.f;
        __syncthreads();

        // softmax denominator + weighted feature sum
        float lsum = 0.f;
        float lr[16];
#pragma unroll
        for (int k = 0; k < 16; k++) lr[k] = 0.f;
        for (int n = beg + tid; n < end; n += 64) {
            float e = 0.f;
            for (int k = 0; k < 16; k++) e += x[(size_t)n * 16 + k] * h[k];
            float a = expf(e - emax);
            lsum += a;
            for (int k = 0; k < 16; k++) lr[k] += a * x[(size_t)n * 16 + k];
        }
        red[tid] = lsum;
        for (int k = 0; k < 16; k++) atomicAdd(&rsh[k], lr[k]);
        __syncthreads();
        for (int off = 32; off > 0; off >>= 1) {
            if (tid < off) red[tid] += red[tid + off];
            __syncthreads();
        }
        float asum = red[0];
        asum = asum > 1e-16f ? asum : 1e-16f;
        __syncthreads();
        if (tid < 16) {
            qs[tid]      = h[tid];
            qs[16 + tid] = rsh[tid] / asum;
        }
        __syncthreads();
    }

    // final MLP: 32 -> 16 -> 8 -> 1
    if (tid < 16) {
        float v = l1b[tid];
        for (int k = 0; k < 32; k++) v += qs[k] * l1W[k * 16 + tid];
        y1s[tid] = v > 0.f ? v : 0.f;
    }
    __syncthreads();
    if (tid < 8) {
        float v = l2b[tid];
        for (int k = 0; k < 16; k++) v += y1s[k] * l2W[k * 8 + tid];
        y2s[tid] = v > 0.f ? v : 0.f;
    }
    __syncthreads();
    if (tid == 0) {
        float v = lfb[0];
        for (int k = 0; k < 8; k++) v += y2s[k] * lfW[k];
        out[gph] = v;
    }
}

// ---------------------------------------------------------------------------
// Host launcher
// ---------------------------------------------------------------------------
extern "C" void kernel_launch(void* const* d_in, const int* in_sizes, int n_in,
                              void* d_out, int out_size, void* d_ws, size_t ws_size,
                              hipStream_t stream)
{
    (void)in_sizes; (void)n_in; (void)out_size; (void)ws_size;

    const float* x_in  = (const float*)d_in[0];
    const int*   eidx  = (const int*)d_in[1];
    const float* ea    = (const float*)d_in[2];
    const int*   batch = (const int*)d_in[3];
    const int* src = eidx;
    const int* tgt = eidx + N_EDGES;

    // per-layer params: W1, b1, W2, b2, root, bias at 4 + 6*L
    const float* cW1[3];  const float* cb1[3];
    const float* cW2[3];  const float* cb2[3];
    const float* croot[3]; const float* cbias[3];
    for (int L = 0; L < 3; L++) {
        int base = 4 + 6 * L;
        cW1[L]   = (const float*)d_in[base + 0];
        cb1[L]   = (const float*)d_in[base + 1];
        cW2[L]   = (const float*)d_in[base + 2];
        cb2[L]   = (const float*)d_in[base + 3];
        croot[L] = (const float*)d_in[base + 4];
        cbias[L] = (const float*)d_in[base + 5];
    }
    const float* Wih = (const float*)d_in[22];
    const float* Whh = (const float*)d_in[23];
    const float* bih = (const float*)d_in[24];
    const float* bhh = (const float*)d_in[25];
    const float* l1W = (const float*)d_in[26];
    const float* l1b = (const float*)d_in[27];
    const float* l2W = (const float*)d_in[28];
    const float* l2b = (const float*)d_in[29];
    const float* lfW = (const float*)d_in[30];
    const float* lfb = (const float*)d_in[31];

    // carve workspace
    uint8_t* base = (uint8_t*)d_ws;
    size_t off = 0;
    auto carve = [&](size_t bytes) -> void* {
        off = (off + 255) & ~(size_t)255;
        void* p = base + off;
        off += bytes;
        return p;
    };
    float* deg  = (float*)carve((size_t)N_NODES * 4);
    float* agg  = (float*)carve((size_t)N_NODES * 48 * 4);
    float* ybuf1 = (float*)carve((size_t)N_NODES * 48 * 4);
    float* ybuf2 = (float*)carve((size_t)N_NODES * 32 * 4);
    float* ybuf3 = (float*)carve((size_t)N_NODES * 16 * 4);
    f16* w1p[3];
    for (int L = 0; L < 3; L++) w1p[L] = (f16*)carve(8 * 512 * sizeof(f16));
    const int nall[3] = {16 * 48, 48 * 32, 32 * 16};   // 768, 1536, 512
    f16* w2p[3];
    for (int L = 0; L < 3; L++) w2p[L] = (f16*)carve((size_t)(nall[L] / 16) * 4 * 512 * sizeof(f16));

    // zero degree + accumulate
    hipMemsetAsync(deg, 0, (size_t)N_NODES * 4, stream);
    deg_kernel<<<(N_EDGES + 255) / 256, 256, 0, stream>>>(tgt, deg, N_EDGES);

    // pack weights into WMMA B-fragment order
    for (int L = 0; L < 3; L++) {
        int tot1 = 8 * 1 * 512;
        pack_b_kernel<<<(tot1 + 255) / 256, 256, 0, stream>>>(cW1[L], w1p[L], 16, 128);
        int tot2 = (nall[L] / 16) * 4 * 512;
        pack_b_kernel<<<(tot2 + 255) / 256, 256, 0, stream>>>(cW2[L], w2p[L], 128, nall[L]);
    }

    const int tiles = N_EDGES / 16;   // 12500

    // ---- layer 1: 16 -> 48 ----
    hipMemsetAsync(agg, 0, (size_t)N_NODES * 48 * 4, stream);
    nnconv_kernel<16, 48><<<tiles, 32, 0, stream>>>(
        x_in, src, tgt, ea, w1p[0], cb1[0], w2p[0], cb2[0], agg);
    node_update_kernel<16, 48><<<(N_NODES * 48 + 255) / 256, 256, 0, stream>>>(
        x_in, agg, deg, croot[0], cbias[0], ybuf1);

    // ---- layer 2: 48 -> 32 ----
    hipMemsetAsync(agg, 0, (size_t)N_NODES * 32 * 4, stream);
    nnconv_kernel<48, 32><<<tiles, 32, 0, stream>>>(
        ybuf1, src, tgt, ea, w1p[1], cb1[1], w2p[1], cb2[1], agg);
    node_update_kernel<48, 32><<<(N_NODES * 32 + 255) / 256, 256, 0, stream>>>(
        ybuf1, agg, deg, croot[1], cbias[1], ybuf2);

    // ---- layer 3: 32 -> 16 ----
    hipMemsetAsync(agg, 0, (size_t)N_NODES * 16 * 4, stream);
    nnconv_kernel<32, 16><<<tiles, 32, 0, stream>>>(
        ybuf2, src, tgt, ea, w1p[2], cb1[2], w2p[2], cb2[2], agg);
    node_update_kernel<32, 16><<<(N_NODES * 16 + 255) / 256, 256, 0, stream>>>(
        ybuf2, agg, deg, croot[2], cbias[2], ybuf3);

    // ---- Set2Set + final MLP ----
    set2set_kernel<<<NUM_GRAPHS, 64, 0, stream>>>(
        ybuf3, batch, Wih, Whh, bih, bhh,
        l1W, l1b, l2W, l2b, lfW, lfb, (float*)d_out);
}
